// Transformer_3221225472124
// MI455X (gfx1250) — matrix operand order repaired and checked by reading, not compiled, not run on hardware
//
#include <hip/hip_runtime.h>
#include <hip/hip_bf16.h>

// ---------------------------------------------------------------------------
// 4-layer transformer fwd on gfx1250 (wave32, WMMA bf16 16x16x32, f32 accum).
// attn has no softmax -> use associativity: attn = q @ (k^T v)  (64x64 per b,h)
// GEMM mainloop: double-buffered LDS filled by GLOBAL_LOAD_ASYNC_TO_LDS_B128
// (ASYNCcnt), one workgroup barrier per K-stage.
// ---------------------------------------------------------------------------

typedef __attribute__((ext_vector_type(16))) __bf16   v16bf;
typedef __attribute__((ext_vector_type(8)))  __bf16   bf16x8;
typedef __attribute__((ext_vector_type(8)))  float    v8f;
typedef __attribute__((ext_vector_type(8)))  unsigned v8u;

union ABFrag { v8u u; v16bf bf; };

#define NB   8
#define NT   1024
#define NH   8
#define NK   64
#define ND   512
#define NFF  2048
#define MTOT (NB * NT)   // 8192 rows

__device__ inline __bf16 f2bf(float f) {
  unsigned u = __float_as_uint(f);
  u += 0x7fffu + ((u >> 16) & 1u);                 // round-to-nearest-even
  unsigned short h = (unsigned short)(u >> 16);
  return __builtin_bit_cast(__bf16, h);
}

__device__ inline float gelu_f(float x) {         // tanh approximation (jax default)
  const float c = 0.7978845608028654f;
  float x3 = x * x * x;
  return 0.5f * x * (1.0f + tanhf(c * (x + 0.044715f * x3)));
}

// --- CDNA5 async global->LDS (tracked by ASYNCcnt) --------------------------
// Generic pointers to __shared__ carry the LDS offset in their low 32 bits.
__device__ inline void async_b128(const __bf16* g, __bf16* lds) {
  unsigned lo = (unsigned)(uintptr_t)lds;
  asm volatile("global_load_async_to_lds_b128 %0, %1, off"
               :: "v"(lo), "v"(g) : "memory");
}
__device__ inline void wait_async0() {
  asm volatile("s_wait_asynccnt 0" ::: "memory");
}

// --- WMMA fragment loaders from LDS (ISA 7.12.2 16-bit layouts) -------------
__device__ inline v16bf load_frag_a(const __bf16* p, int half) {
  ABFrag f;
#pragma unroll
  for (int j = 0; j < 8; ++j)
    f.u[j] = *(const unsigned*)(p + ((j >> 2) << 4) + (half << 3) + ((j & 3) << 1));
  return f.bf;
}
__device__ inline v16bf load_frag_b(const __bf16* p, int half) {
  ABFrag f;
#pragma unroll
  for (int j = 0; j < 8; ++j)
    f.u[j] = *(const unsigned*)(p + ((j >> 2) << 4) + (half << 3) + ((j & 3) << 1));
  return f.bf;
}

__device__ inline v8f wmma_bf16(v16bf a, v16bf b, v8f c) {
  return __builtin_amdgcn_wmma_f32_16x16x32_bf16(false, a, false, b, (short)0, c,
                                                 false, false);
}

// --- weight transpose + bf16 convert: w[Kd][N] -> wT[N][Kd] -----------------
__global__ void transpose_cvt_kernel(const float* __restrict__ w,
                                     __bf16* __restrict__ wT, int Kd, int N) {
  int idx = blockIdx.x * blockDim.x + threadIdx.x;
  if (idx >= Kd * N) return;
  int k = idx / N, n = idx - k * N;
  wT[(size_t)n * Kd + k] = f2bf(w[idx]);
}

// --- layernorm over D=512, one wave per row, output bf16 --------------------
__global__ void __launch_bounds__(128)
layernorm_kernel(const float* __restrict__ x, const float* __restrict__ sc,
                 const float* __restrict__ bi, __bf16* __restrict__ out) {
  const int row  = blockIdx.x * 4 + (threadIdx.x >> 5);
  const int lane = threadIdx.x & 31;
  const float* p = x + (size_t)row * ND;
  float v[16], s = 0.0f;
#pragma unroll
  for (int i = 0; i < 16; ++i) { v[i] = p[lane + i * 32]; s += v[i]; }
  s += __shfl_xor(s, 16); s += __shfl_xor(s, 8); s += __shfl_xor(s, 4);
  s += __shfl_xor(s, 2);  s += __shfl_xor(s, 1);
  const float mean = s * (1.0f / ND);
  float q = 0.0f;
#pragma unroll
  for (int i = 0; i < 16; ++i) { float d = v[i] - mean; q += d * d; }
  q += __shfl_xor(q, 16); q += __shfl_xor(q, 8); q += __shfl_xor(q, 4);
  q += __shfl_xor(q, 2);  q += __shfl_xor(q, 1);
  const float inv = rsqrtf(q * (1.0f / ND) + 1e-6f);
#pragma unroll
  for (int i = 0; i < 16; ++i) {
    int c = lane + i * 32;
    out[(size_t)row * ND + c] = f2bf((v[i] - mean) * inv * sc[c] + bi[c]);
  }
}

// --- main GEMM: out[M][N] = A[M][Kd] @ WT[N][Kd]^T, fused epilogues ---------
// MODE 0: +bias, store bf16                          (v projection)
// MODE 1: +bias, per-head sum-normalize, store bf16  (q,k projections)
// MODE 2: h = clip(h + 0.9*(acc+bias))               (wo / w2 residual)
// MODE 3: store bf16 gelu(acc+bias)                  (w1)
// Tiles: BM=BN=128, BK=32, double-buffered async LDS fill, 1 barrier/stage.
template <int MODE>
__global__ void __launch_bounds__(256)
gemm_bf16_kernel(const __bf16* __restrict__ A, const __bf16* __restrict__ WT,
                 const float* __restrict__ bias, int M, int N, int Kd,
                 __bf16* __restrict__ outb, float* __restrict__ hio) {
  __shared__ __bf16 sA[2][128][40];   // row = 80B (16B-aligned, low-conflict)
  __shared__ __bf16 sB[2][128][40];
  const int tid  = threadIdx.x;
  const int lane = tid & 31, wave = tid >> 5;
  const int half = lane >> 4, lr = lane & 15;
  const int wm = wave & 3, wn = wave >> 2;      // 4x2 wave grid
  const int m0 = blockIdx.y * 128, n0 = blockIdx.x * 128;

  auto issue = [&](int k0, int buf) {
#pragma unroll
    for (int i = 0; i < 2; ++i) {
      int idx = tid + i * 256;                  // 0..511
      int r = idx >> 2, c = idx & 3;
      async_b128(A  + (size_t)(m0 + r) * Kd + k0 + c * 8, &sA[buf][r][c * 8]);
      async_b128(WT + (size_t)(n0 + r) * Kd + k0 + c * 8, &sB[buf][r][c * 8]);
    }
  };

  v8f acc[2][4] = {};
  const int nk = Kd / 32;
  issue(0, 0);
  for (int s = 0; s < nk; ++s) {
    wait_async0();          // my stage-s async fills landed
    __syncthreads();        // everyone's landed; everyone done reading buf s^1
    if (s + 1 < nk) issue((s + 1) * 32, (s + 1) & 1);
    const int buf = s & 1;
    v16bf af[2], bf[4];
#pragma unroll
    for (int mi = 0; mi < 2; ++mi)
      af[mi] = load_frag_a(&sA[buf][wm * 32 + mi * 16 + lr][0], half);
#pragma unroll
    for (int ni = 0; ni < 4; ++ni)
      bf[ni] = load_frag_b(&sB[buf][wn * 64 + ni * 16 + lr][0], half);
#pragma unroll
    for (int mi = 0; mi < 2; ++mi)
#pragma unroll
      for (int ni = 0; ni < 4; ++ni)
        acc[mi][ni] = wmma_bf16(af[mi], bf[ni], acc[mi][ni]);
  }

  // epilogue: C layout -> VGPR r: rows m = r + half*8, col = lr within tile
#pragma unroll
  for (int mi = 0; mi < 2; ++mi) {
#pragma unroll
    for (int r = 0; r < 8; ++r) {
      const int m = m0 + wm * 32 + mi * 16 + r + half * 8;
      if (MODE == 1) {
        float v[4], s = 0.0f;
#pragma unroll
        for (int ni = 0; ni < 4; ++ni) {
          int n = n0 + wn * 64 + ni * 16 + lr;
          v[ni] = acc[mi][ni][r] + bias[n];
          s += v[ni];
        }
        // head's 64-col row sum: 4 tiles + 16-lane half-wave reduce
        s += __shfl_xor(s, 1); s += __shfl_xor(s, 2);
        s += __shfl_xor(s, 4); s += __shfl_xor(s, 8);
        const float den = s + 1e-6f;
#pragma unroll
        for (int ni = 0; ni < 4; ++ni) {
          int n = n0 + wn * 64 + ni * 16 + lr;
          outb[(size_t)m * N + n] = f2bf(v[ni] / den);
        }
      } else {
#pragma unroll
        for (int ni = 0; ni < 4; ++ni) {
          int n = n0 + wn * 64 + ni * 16 + lr;
          float val = acc[mi][ni][r] + bias[n];
          size_t o = (size_t)m * N + n;
          if (MODE == 0) {
            outb[o] = f2bf(val);
          } else if (MODE == 2) {
            float xo = hio[o] + 0.9f * val;
            xo = fminf(5.0f, fmaxf(-5.0f, xo));
            hio[o] = xo;
          } else {
            outb[o] = f2bf(gelu_f(val));
          }
        }
      }
    }
  }
}

// --- S[b,h] = k^T v  (64x64, K = T-1 = 1023, one WG per (b,h)) --------------
__global__ void __launch_bounds__(256)
kv_outer_kernel(const __bf16* __restrict__ kb, const __bf16* __restrict__ vb,
                __bf16* __restrict__ Sb) {
  const int b = blockIdx.x >> 3, h = blockIdx.x & 7;
  __shared__ __bf16 sK[64][40];   // [e][t]  (contraction t contiguous)
  __shared__ __bf16 sV[64][40];   // [d][t]
  const int tid = threadIdx.x, lane = tid & 31, wave = tid >> 5;
  const int half = lane >> 4, lr = lane & 15;
  const int mt = wave >> 1;
  const __bf16 z = f2bf(0.0f);
  v8f acc[2] = {};
  for (int t0 = 0; t0 < NT - 1; t0 += 32) {
    __syncthreads();
    {
      // one b128 load per lane per matrix, scatter-transpose into LDS
      const int tl = tid >> 3;                  // 0..31
      const int c  = tid & 7;                   // k chunk
      const int t  = t0 + tl;
      const bool ok = t < NT - 1;               // mask t = T-1
      bf16x8 k8, v8;
#pragma unroll
      for (int e = 0; e < 8; ++e) { k8[e] = z; v8[e] = z; }
      if (ok) {
        size_t g = ((size_t)(b * NT + t)) * ND + h * NK + c * 8;
        k8 = *(const bf16x8*)(kb + g);
        v8 = *(const bf16x8*)(vb + g);
      }
#pragma unroll
      for (int e = 0; e < 8; ++e) {
        sK[c * 8 + e][tl] = k8[e];
        sV[c * 8 + e][tl] = v8[e];
      }
    }
    __syncthreads();
    v16bf a = load_frag_a(&sK[mt * 16 + lr][0], half);
#pragma unroll
    for (int ni = 0; ni < 2; ++ni) {
      int nt = (wave & 1) * 2 + ni;
      v16bf bb = load_frag_b(&sV[nt * 16 + lr][0], half);
      acc[ni] = wmma_bf16(a, bb, acc[ni]);
    }
  }
#pragma unroll
  for (int ni = 0; ni < 2; ++ni) {
    int nt = (wave & 1) * 2 + ni;
#pragma unroll
    for (int r = 0; r < 8; ++r) {
      int m = mt * 16 + r + half * 8;
      int n = nt * 16 + lr;
      Sb[((size_t)blockIdx.x * 64 + m) * 64 + n] = f2bf(acc[ni][r]);
    }
  }
}

// --- attn[b,t,h,:] = q[b,t,h,:] @ S[b,h]  (M=128 tile, N=64, K=64) ----------
__global__ void __launch_bounds__(256)
attn_apply_kernel(const __bf16* __restrict__ qb, const __bf16* __restrict__ Sb,
                  __bf16* __restrict__ attnb) {
  const int bh = blockIdx.x, b = bh >> 3, h = bh & 7;
  const int t0 = blockIdx.y * 128;
  __shared__ __bf16 sQ[128][72];  // [t][e], row 144B
  __shared__ __bf16 sS[64][72];   // [d][e] (transposed S)
  const int tid = threadIdx.x, lane = tid & 31, wave = tid >> 5;
  const int half = lane >> 4, lr = lane & 15;
#pragma unroll
  for (int i = 0; i < 4; ++i) {
    int idx = tid + i * 256;                    // 0..1023
    int r = idx >> 3, c = idx & 7;
    *(bf16x8*)&sQ[r][c * 8] =
        *(const bf16x8*)(qb + ((size_t)(b * NT + t0 + r)) * ND + h * NK + c * 8);
  }
#pragma unroll
  for (int i = 0; i < 16; ++i) {
    int idx = tid + i * 256;                    // 0..4095
    int e = idx >> 6, d = idx & 63;
    sS[d][e] = Sb[(size_t)bh * 4096 + idx];
  }
  __syncthreads();
  v8f acc[4] = {};
#pragma unroll
  for (int ks = 0; ks < 2; ++ks) {
    v16bf a = load_frag_a(&sQ[wave * 16 + lr][ks * 32], half);
#pragma unroll
    for (int ni = 0; ni < 4; ++ni) {
      v16bf bb = load_frag_b(&sS[ni * 16 + lr][ks * 32], half);
      acc[ni] = wmma_bf16(a, bb, acc[ni]);
    }
  }
#pragma unroll
  for (int ni = 0; ni < 4; ++ni)
#pragma unroll
    for (int r = 0; r < 8; ++r) {
      int m = wave * 16 + r + half * 8;
      int n = ni * 16 + lr;
      attnb[((size_t)(b * NT + t0 + m)) * ND + h * NK + n] = f2bf(acc[ni][r]);
    }
}

// ---------------------------------------------------------------------------
extern "C" void kernel_launch(void* const* d_in, const int* in_sizes, int n_in,
                              void* d_out, int out_size, void* d_ws, size_t ws_size,
                              hipStream_t stream) {
  (void)in_sizes; (void)n_in; (void)out_size; (void)ws_size;
  const float* x    = (const float*)d_in[0];
  const float* wq   = (const float*)d_in[1];
  const float* bq   = (const float*)d_in[2];
  const float* wk   = (const float*)d_in[3];
  const float* bk   = (const float*)d_in[4];
  const float* wv   = (const float*)d_in[5];
  const float* bv   = (const float*)d_in[6];
  const float* wo   = (const float*)d_in[7];
  const float* bo   = (const float*)d_in[8];
  const float* ln1s = (const float*)d_in[9];
  const float* ln1b = (const float*)d_in[10];
  const float* ln2s = (const float*)d_in[11];
  const float* ln2b = (const float*)d_in[12];
  const float* w1   = (const float*)d_in[13];
  const float* b1   = (const float*)d_in[14];
  const float* w2   = (const float*)d_in[15];
  const float* b2   = (const float*)d_in[16];

  char* base = (char*)d_ws;
  size_t off = 0;
  auto take = [&](size_t bytes) -> char* {
    char* p = base + off;
    off += (bytes + 255) & ~(size_t)255;
    return p;
  };
  float*  h     = (float*) take((size_t)MTOT * ND * 4);
  __bf16* hnb   = (__bf16*)take((size_t)MTOT * ND * 2);
  __bf16* qb    = (__bf16*)take((size_t)MTOT * ND * 2);
  __bf16* kb    = (__bf16*)take((size_t)MTOT * ND * 2);
  __bf16* vb    = (__bf16*)take((size_t)MTOT * ND * 2);
  __bf16* attnb = (__bf16*)take((size_t)MTOT * ND * 2);
  __bf16* Sb    = (__bf16*)take((size_t)NB * NH * NK * NK * 2);
  __bf16* gelub = (__bf16*)take((size_t)MTOT * NFF * 2);
  __bf16* wqT   = (__bf16*)take((size_t)ND * ND * 2);
  __bf16* wkT   = (__bf16*)take((size_t)ND * ND * 2);
  __bf16* wvT   = (__bf16*)take((size_t)ND * ND * 2);
  __bf16* woT   = (__bf16*)take((size_t)ND * ND * 2);
  __bf16* w1T   = (__bf16*)take((size_t)ND * NFF * 2);
  __bf16* w2T   = (__bf16*)take((size_t)NFF * ND * 2);

  hipMemcpyAsync(h, x, (size_t)MTOT * ND * 4, hipMemcpyDeviceToDevice, stream);

  transpose_cvt_kernel<<<(ND * ND + 255) / 256, 256, 0, stream>>>(wq, wqT, ND, ND);
  transpose_cvt_kernel<<<(ND * ND + 255) / 256, 256, 0, stream>>>(wk, wkT, ND, ND);
  transpose_cvt_kernel<<<(ND * ND + 255) / 256, 256, 0, stream>>>(wv, wvT, ND, ND);
  transpose_cvt_kernel<<<(ND * ND + 255) / 256, 256, 0, stream>>>(wo, woT, ND, ND);
  transpose_cvt_kernel<<<(ND * NFF + 255) / 256, 256, 0, stream>>>(w1, w1T, ND, NFF);
  transpose_cvt_kernel<<<(NFF * ND + 255) / 256, 256, 0, stream>>>(w2, w2T, NFF, ND);

  const dim3 blk(256);
  const dim3 gD(ND / 128, MTOT / 128);     // 4 x 64
  const dim3 gFF(NFF / 128, MTOT / 128);   // 16 x 64

  for (int layer = 0; layer < 4; ++layer) {
    layernorm_kernel<<<MTOT / 4, 128, 0, stream>>>(h, ln1s, ln1b, hnb);
    gemm_bf16_kernel<1><<<gD, blk, 0, stream>>>(hnb, wqT, bq, MTOT, ND, ND, qb, nullptr);
    gemm_bf16_kernel<1><<<gD, blk, 0, stream>>>(hnb, wkT, bk, MTOT, ND, ND, kb, nullptr);
    gemm_bf16_kernel<0><<<gD, blk, 0, stream>>>(hnb, wvT, bv, MTOT, ND, ND, vb, nullptr);
    kv_outer_kernel<<<NB * NH, blk, 0, stream>>>(kb, vb, Sb);
    attn_apply_kernel<<<dim3(NB * NH, NT / 128), blk, 0, stream>>>(qb, Sb, attnb);
    gemm_bf16_kernel<2><<<gD, blk, 0, stream>>>(attnb, woT, bo, MTOT, ND, ND, nullptr, h);
    layernorm_kernel<<<MTOT / 4, 128, 0, stream>>>(h, ln2s, ln2b, hnb);
    gemm_bf16_kernel<3><<<gFF, blk, 0, stream>>>(hnb, w1T, b1, MTOT, NFF, ND, gelub, nullptr);
    gemm_bf16_kernel<2><<<gD, blk, 0, stream>>>(gelub, w2T, b2, MTOT, ND, NFF, nullptr, h);
  }

  hipMemcpyAsync(d_out, h, (size_t)MTOT * ND * 4, hipMemcpyDeviceToDevice, stream);
}
